// MultiAttentionHead_25572235281071
// MI455X (gfx1250) — compile-verified, matrix operand order checked by reading, not run on hardware
//
#include <hip/hip_runtime.h>
#include <hip/hip_bf16.h>

#define BB 2
#define TT 2048
#define CC 1024
#define HH 16
#define HS 64

typedef __attribute__((ext_vector_type(16))) __bf16        v16bf;
typedef __attribute__((ext_vector_type(8)))  float         v8f;
typedef __attribute__((ext_vector_type(8)))  unsigned int  v8u;
typedef __attribute__((ext_vector_type(4)))  float         v4f;
typedef __attribute__((ext_vector_type(4)))  unsigned short v4h;

// ---------- helpers ----------

static __device__ __forceinline__ unsigned short f2bf(float f) {
  unsigned int u = __builtin_bit_cast(unsigned int, f);
  u += 0x7FFFu + ((u >> 16) & 1u);   // round-to-nearest-even
  return (unsigned short)(u >> 16);
}

// Load a 16x32 bf16 fragment (A layout; B layout is identical when the
// operand is stored [n][k] row-major).  rowbase -> element [row0][0].
// Lane L: row = L&15, half = L>>4; VGPR i holds packed K-pair at
// k = (i>>2)*16 + half*8 + (i&3)*2  (CDNA5 ISA 7.12.2 layout).
static __device__ __forceinline__ v16bf load_frag(const unsigned short* rowbase,
                                                  int stride, int k0, int lane) {
  const int sel = lane & 15, half = lane >> 4;
  const unsigned int* p =
      (const unsigned int*)(rowbase + (size_t)sel * stride + k0);
  v8u u;
#pragma unroll
  for (int i = 0; i < 8; ++i) {
    int k = ((i >> 2) << 4) + (half << 3) + ((i & 3) << 1);
    u[i] = p[k >> 1];
  }
  return __builtin_bit_cast(v16bf, u);
}

static __device__ __forceinline__ v8f wmma_bf16(v16bf a, v16bf b, v8f c) {
  return __builtin_amdgcn_wmma_f32_16x16x32_bf16(false, a, false, b, (short)0, c,
                                                 false, false);
}

// ---------- conversion kernels ----------

// float4 -> 4 x bf16 per thread (n must be a multiple of 4; all ours are).
__global__ void cvt_bf16_v4(const float* __restrict__ in,
                            unsigned short* __restrict__ out, int n4) {
  int i = blockIdx.x * blockDim.x + threadIdx.x;
  if (i >= n4) return;
  v4f x = ((const v4f*)in)[i];
  v4h y;
#pragma unroll
  for (int j = 0; j < 4; ++j) y[j] = f2bf(x[j]);
  ((v4h*)out)[i] = y;
}

// in[k][n] fp32 -> out[n][k] bf16, rows fixed to CC=1024 (shift/mask only).
__global__ void cvt_bf16_tc(const float* __restrict__ in,
                            unsigned short* __restrict__ out, int cols, int total) {
  int i = blockIdx.x * blockDim.x + threadIdx.x;
  if (i >= total) return;
  int k = i & (CC - 1);
  int n = i >> 10;
  out[((size_t)n << 10) + k] = f2bf(in[(size_t)k * cols + n]);
}

// ---------- QKV GEMM:  X[4096x1024] @ Wqkv -> per-head Q,K,V ----------
// Register-blocked 32M x 64N per wave: 12 b128 loads feed 8 WMMAs per K-step.

__global__ void __launch_bounds__(256) qkv_gemm(
    const unsigned short* __restrict__ Xb,  // [M][K] bf16
    const unsigned short* __restrict__ Wt,  // [N=3072][K=1024] bf16 (W^T)
    unsigned short* __restrict__ Qh,        // [B,H,T,HS]
    unsigned short* __restrict__ Kh,        // [B,H,T,HS]
    unsigned short* __restrict__ Vt)        // [B,H,HS,T]  (transposed)
{
  const int lane = threadIdx.x & 31;
  const int wave = threadIdx.x >> 5;
  const int tile = blockIdx.x * 8 + wave;
  const int tilesM = (BB * TT) / 32;        // 128
  const int mt = tile % tilesM;
  const int nt = tile / tilesM;             // 0..47
  const int m0 = mt * 32, n0 = nt * 64;
  const unsigned short* Arow0 = Xb + (size_t)m0 * CC;
  const unsigned short* Arow1 = Arow0 + (size_t)16 * CC;
  const unsigned short* Brow0 = Wt + (size_t)n0 * CC;
  const int sel = lane & 15, half = lane >> 4;

  // one-shot panel warm-up (WGP-scope prefetch, outside the hot loop)
  __builtin_prefetch(Arow0 + (size_t)sel * CC, 0, 3);
  __builtin_prefetch(Arow1 + (size_t)sel * CC, 0, 3);
  __builtin_prefetch(Brow0 + (size_t)(half * 48 + sel) * CC, 0, 3);

  v8f acc[2][4] = {};
#pragma unroll 2
  for (int k0 = 0; k0 < CC; k0 += 32) {
    v16bf a0 = load_frag(Arow0, CC, k0, lane);
    v16bf a1 = load_frag(Arow1, CC, k0, lane);
#pragma unroll
    for (int ni = 0; ni < 4; ++ni) {
      v16bf b = load_frag(Brow0 + (size_t)(ni * 16) * CC, CC, k0, lane);
      acc[0][ni] = wmma_bf16(a0, b, acc[0][ni]);
      acc[1][ni] = wmma_bf16(a1, b, acc[1][ni]);
    }
  }

#pragma unroll
  for (int ni = 0; ni < 4; ++ni) {
    const int ng = n0 + ni * 16 + sel;
    const int section = ng >> 10;           // 0=K, 1=Q, 2=V (module order!)
    const int c = ng & (CC - 1);
    const int head = c >> 6;
    const int d = c & (HS - 1);
#pragma unroll
    for (int mi = 0; mi < 2; ++mi) {
#pragma unroll
      for (int r = 0; r < 8; ++r) {
        int m = m0 + mi * 16 + r + (half << 3);
        int b_ = m >> 11;                   // m / T
        int t = m & (TT - 1);
        unsigned short v = f2bf(acc[mi][ni][r]);
        if (section == 0)
          Kh[(((size_t)b_ * HH + head) * TT + t) * HS + d] = v;
        else if (section == 1)
          Qh[(((size_t)b_ * HH + head) * TT + t) * HS + d] = v;
        else
          Vt[(((size_t)b_ * HH + head) * HS + d) * TT + t] = v;
      }
    }
  }
}

// ---------- Flash attention: one wave per (b, h, 16-query tile) ----------

__global__ void __launch_bounds__(32) flash_attn(
    const unsigned short* __restrict__ Qh,
    const unsigned short* __restrict__ Kh,
    const unsigned short* __restrict__ Vt,
    unsigned short* __restrict__ attn)      // [B,T,C] bf16
{
  __shared__ unsigned short pbuf[16 * 32];  // P transpose staging (1 KB)
  const int lane = threadIdx.x & 31;
  const int qi = blockIdx.x & 127;
  const int h  = (blockIdx.x >> 7) & 15;
  const int b  = blockIdx.x >> 11;
  const size_t bh = (size_t)b * HH + h;
  const unsigned short* Qrow  = Qh + (bh * TT + (size_t)qi * 16) * HS;
  const unsigned short* Kbase = Kh + bh * TT * HS;
  const unsigned short* Vbase = Vt + bh * HS * TT;
  const int sel = lane & 15, half = lane >> 4;

  const v16bf aq0 = load_frag(Qrow, HS, 0, lane);   // Q tile, K-dim 0..31
  const v16bf aq1 = load_frag(Qrow, HS, 32, lane);  // Q tile, K-dim 32..63

  float mrow[8], lrow[8];
  v8f accO[4] = {};
#pragma unroll
  for (int r = 0; r < 8; ++r) { mrow[r] = -__builtin_inff(); lrow[r] = 0.f; }

  const float scale = 0.03125f;             // C^-0.5 = 1024^-0.5
  const int nkb = (qi >> 1) + 1;            // 32-key blocks under causal limit
  for (int kb = 0; kb < nkb; ++kb) {
    const int key0 = kb * 32;
    // S = Q @ K^T   (two 16-key column tiles, K-dim 64 = 2 WMMAs each)
    v8f s0 = {}, s1 = {};
    s0 = wmma_bf16(aq0, load_frag(Kbase + (size_t)key0 * HS, HS, 0, lane), s0);
    s0 = wmma_bf16(aq1, load_frag(Kbase + (size_t)key0 * HS, HS, 32, lane), s0);
    s1 = wmma_bf16(aq0, load_frag(Kbase + (size_t)(key0 + 16) * HS, HS, 0, lane), s1);
    s1 = wmma_bf16(aq1, load_frag(Kbase + (size_t)(key0 + 16) * HS, HS, 32, lane), s1);

    float p0[8], p1[8];
#pragma unroll
    for (int r = 0; r < 8; ++r) {
      const int row = qi * 16 + r + (half << 3);
      float x0 = (key0 + sel      > row) ? -__builtin_inff() : s0[r] * scale;
      float x1 = (key0 + 16 + sel > row) ? -__builtin_inff() : s1[r] * scale;
      // row max across the 16-lane half (rows live within one half)
      float v = fmaxf(x0, x1);
#pragma unroll
      for (int off = 1; off < 16; off <<= 1) v = fmaxf(v, __shfl_xor(v, off, 32));
      const float mnew = fmaxf(mrow[r], v);
      const float corr = __expf(mrow[r] - mnew);
      const float e0 = __expf(x0 - mnew);
      const float e1 = __expf(x1 - mnew);
      float s = e0 + e1;
#pragma unroll
      for (int off = 1; off < 16; off <<= 1) s += __shfl_xor(s, off, 32);
      lrow[r] = lrow[r] * corr + s;
      mrow[r] = mnew;
#pragma unroll
      for (int g = 0; g < 4; ++g) accO[g][r] *= corr;
      p0[r] = e0; p1[r] = e1;
    }

    // transpose P (C-layout -> A-layout) through LDS
#pragma unroll
    for (int r = 0; r < 8; ++r) {
      const int rl = r + (half << 3);
      pbuf[rl * 32 + sel]      = f2bf(p0[r]);
      pbuf[rl * 32 + 16 + sel] = f2bf(p1[r]);
    }
    __syncthreads();
    const v16bf pa = load_frag(pbuf, 32, 0, lane);
    __syncthreads();

    // O += P @ V   (V stored transposed [d][t] -> contiguous key pairs)
#pragma unroll
    for (int g = 0; g < 4; ++g) {
      v16bf bv = load_frag(Vbase + (size_t)(g * 16) * TT + key0, TT, 0, lane);
      accO[g] = wmma_bf16(pa, bv, accO[g]);
    }
  }

  float inv[8];
#pragma unroll
  for (int r = 0; r < 8; ++r) inv[r] = 1.f / lrow[r];
#pragma unroll
  for (int g = 0; g < 4; ++g)
#pragma unroll
    for (int r = 0; r < 8; ++r) {
      const int row = qi * 16 + r + (half << 3);
      const int col = h * HS + g * 16 + sel;
      attn[((size_t)b * TT + row) * CC + col] = f2bf(accO[g][r] * inv[r]);
    }
}

// ---------- Projection GEMM: attn[4096x1024] @ Wproj -> fp32 out ----------
// Same 32M x 64N register blocking.

__global__ void __launch_bounds__(256) proj_gemm(
    const unsigned short* __restrict__ A,   // [M][K] bf16
    const unsigned short* __restrict__ Wt,  // [N][K] bf16 (W^T)
    float* __restrict__ out)                // [M][N] fp32
{
  const int lane = threadIdx.x & 31;
  const int wave = threadIdx.x >> 5;
  const int tile = blockIdx.x * 8 + wave;
  const int tilesM = (BB * TT) / 32;        // 128
  const int mt = tile % tilesM;
  const int nt = tile / tilesM;             // 0..15
  const int m0 = mt * 32, n0 = nt * 64;
  const unsigned short* Arow0 = A + (size_t)m0 * CC;
  const unsigned short* Arow1 = Arow0 + (size_t)16 * CC;
  const unsigned short* Brow0 = Wt + (size_t)n0 * CC;
  const int sel = lane & 15, half = lane >> 4;

  __builtin_prefetch(Arow0 + (size_t)sel * CC, 0, 3);
  __builtin_prefetch(Arow1 + (size_t)sel * CC, 0, 3);
  __builtin_prefetch(Brow0 + (size_t)(half * 48 + sel) * CC, 0, 3);

  v8f acc[2][4] = {};
#pragma unroll 2
  for (int k0 = 0; k0 < CC; k0 += 32) {
    v16bf a0 = load_frag(Arow0, CC, k0, lane);
    v16bf a1 = load_frag(Arow1, CC, k0, lane);
#pragma unroll
    for (int ni = 0; ni < 4; ++ni) {
      v16bf b = load_frag(Brow0 + (size_t)(ni * 16) * CC, CC, k0, lane);
      acc[0][ni] = wmma_bf16(a0, b, acc[0][ni]);
      acc[1][ni] = wmma_bf16(a1, b, acc[1][ni]);
    }
  }

#pragma unroll
  for (int ni = 0; ni < 4; ++ni)
#pragma unroll
    for (int mi = 0; mi < 2; ++mi)
#pragma unroll
      for (int r = 0; r < 8; ++r) {
        const int m = m0 + mi * 16 + r + (half << 3);
        out[(size_t)m * CC + n0 + ni * 16 + sel] = acc[mi][ni][r];
      }
}

// ---------- launch ----------

extern "C" void kernel_launch(void* const* d_in, const int* in_sizes, int n_in,
                              void* d_out, int out_size, void* d_ws, size_t ws_size,
                              hipStream_t stream) {
  const float* X     = (const float*)d_in[0];   // [B,T,C]
  const float* Wqkv  = (const float*)d_in[1];   // [C,3C]
  const float* Wproj = (const float*)d_in[2];   // [C,C]
  float* out = (float*)d_out;

  char* ws = (char*)d_ws;
  unsigned short* Xb   = (unsigned short*)(ws);                       //  8 MB
  unsigned short* Wqt  = (unsigned short*)(ws + (size_t)( 8u << 20)); //  6 MB
  unsigned short* Wpt  = (unsigned short*)(ws + (size_t)(14u << 20)); //  2 MB
  unsigned short* Qh   = (unsigned short*)(ws + (size_t)(16u << 20)); //  8 MB
  unsigned short* Kh   = (unsigned short*)(ws + (size_t)(24u << 20)); //  8 MB
  unsigned short* Vt   = (unsigned short*)(ws + (size_t)(32u << 20)); //  8 MB
  unsigned short* attn = (unsigned short*)(ws + (size_t)(40u << 20)); //  8 MB

  const int nX4 = BB * TT * CC / 4;
  cvt_bf16_v4<<<dim3((nX4 + 255) / 256), dim3(256), 0, stream>>>(X, Xb, nX4);
  cvt_bf16_tc<<<dim3((CC * 3 * CC + 255) / 256), dim3(256), 0, stream>>>(
      Wqkv, Wqt, 3 * CC, CC * 3 * CC);
  cvt_bf16_tc<<<dim3((CC * CC + 255) / 256), dim3(256), 0, stream>>>(
      Wproj, Wpt, CC, CC * CC);

  // 32x64 tiles: qkv = (4096/32)*(3072/64) = 6144 waves; proj = 2048 waves.
  qkv_gemm<<<dim3(128 * 48 / 8), dim3(256), 0, stream>>>(Xb, Wqt, Qh, Kh, Vt);
  flash_attn<<<dim3(BB * HH * (TT / 16)), dim3(32), 0, stream>>>(Qh, Kh, Vt, attn);
  proj_gemm<<<dim3(128 * 16 / 8), dim3(256), 0, stream>>>(attn, Wpt, out);
}
